// ODEFunc_17617955848577
// MI455X (gfx1250) — matrix operand order repaired and checked by reading, not compiled, not run on hardware
//
#include <hip/hip_runtime.h>
#include <hip/hip_bf16.h>

typedef __attribute__((ext_vector_type(16))) _Float16 v16h;
typedef __attribute__((ext_vector_type(8)))  float    v8f;

#define NKNN 10

// One particle per lane. Per wave (32 particles):
//   - 10 neighbor-MLP tiles + 4 wall-MLP tiles, each one v_wmma_f32_16x16x32_f16
//   - A = block-diag(Wb^T, Wb^T) packed per the 16-bit A VGPR layout (constant)
//   - B column s / K-half h  = layer-1 activations of particle 16h+s (lane-local)
//   - C VGPR r in lane l     = hidden unit r of lane l's sample (lane-local layer 3)
__global__ __launch_bounds__(256) void odefunc_kernel(
    const float* __restrict__ y,
    const float* __restrict__ w1a, const float* __restrict__ b1a,
    const float* __restrict__ w1b, const float* __restrict__ b1b,
    const float* __restrict__ w2a, const float* __restrict__ b2a,
    const float* __restrict__ w2b, const float* __restrict__ b2b,
    const float* __restrict__ w2c, const float* __restrict__ b2c,
    const float* __restrict__ w3a, const float* __restrict__ b3a,
    const float* __restrict__ w3b, const float* __restrict__ b3b,
    const float* __restrict__ w3c, const float* __restrict__ b3c,
    float* __restrict__ out, int n)
{
    const int tid  = blockIdx.x * blockDim.x + threadIdx.x;
    const int pid  = (tid < n) ? tid : 0;       // keep EXEC full for WMMA
    const int lane = threadIdx.x & 31;
    const int h    = lane >> 4;                 // K-half / sample group
    const int m    = lane & 15;                 // A-matrix row (M)

    // ---- load state: 24 useful floats, row stride 44 floats (16B aligned) ----
    const float4* y4 = reinterpret_cast<const float4*>(y);
    const size_t  row = (size_t)pid * 11;
    float4 r0 = y4[row + 0];                    // x0, x1, vx, vy
    float ux[2 * NKNN];
    #pragma unroll
    for (int q = 0; q < 5; ++q) {
        float4 u = y4[row + 1 + q];
        ux[4*q+0] = u.x; ux[4*q+1] = u.y; ux[4*q+2] = u.z; ux[4*q+3] = u.w;
    }
    const float x0 = r0.x, x1 = r0.y, vx = r0.z, vy = r0.w;

    // ---- build constant A matrices: block-diag(w3b^T,w3b^T), (w2b^T,w2b^T) ----
    // A packed in contiguous k order (B matches): lane(h,m), element j -> K = 16h + j
    v16h a_nn, a_ww;
    #pragma unroll
    for (int j = 0; j < 16; ++j) {
        int K = 16*h + j;
        bool xblk = (m < 8) && (K < 16);        // rows 0-7  x K 0-15  -> w*b^T
        bool yblk = (m >= 8) && (K >= 16);      // rows 8-15 x K 16-31 -> w*b^T
        int  kk   = xblk ? K : (K - 16);
        int  mm   = xblk ? m : (m - 8);
        float sel = (xblk || yblk) ? 1.f : 0.f;
        int idx = kk * 8 + mm;                  // w*b is (16,8) row-major
        idx = idx < 0 ? 0 : (idx > 127 ? 127 : idx);
        a_nn[j] = (_Float16)(w3b[idx] * sel);
        a_ww[j] = (_Float16)(w2b[idx] * sel);
    }

    // ---- fused scalar->16 -> WMMA(16->8) -> lane-local 8->1 ----
    auto mlp = [&](float in, const float* wa, const float* ba,
                   const float* bb, const float* wc, float bc0,
                   const v16h& A) -> float {
        v16h b;
        #pragma unroll
        for (int j = 0; j < 16; ++j) {          // layer 1 (lane-local)
            float t = fmaf(in, wa[j], ba[j]);
            b[j] = (_Float16)fmaxf(t, 0.f);
        }
        v8f c = {};
        c = __builtin_amdgcn_wmma_f32_16x16x32_f16(
                false, A, false, b, (short)0, c, false, false);
        float acc = bc0;                        // layer 3 (lane-local)
        #pragma unroll
        for (int r = 0; r < 8; ++r) {
            float t = fmaxf(c[r] + bb[r], 0.f);
            acc = fmaf(t, wc[r], acc);
        }
        return acc;
    };

    // ---- f3: neighbor interaction net over 10 distances ----
    const float b3c0 = b3c[0];
    float f3x = 0.f, f3y = 0.f;
    #pragma unroll
    for (int k = 0; k < NKNN; ++k) {
        float ax = ux[2*k], ay = ux[2*k+1];
        float s2   = fmaf(ax, ax, ay * ay);
        float invd = rsqrtf(s2);                // matches ref's ux/d (NaN on 0 too)
        float d    = s2 * invd;                 // sqrt(s2)
        float o    = mlp(d, w3a, b3a, b3b, w3c, b3c0, a_nn);
        float g    = o * invd;
        f3x = fmaf(-g, ax, f3x);
        f3y = fmaf(-g, ay, f3y);
    }

    // ---- f2: wall net over 4 distances ----
    const float b2c0 = b2c[0];
    float rightdist = ((x1 < 1.f) && (x1 > -1.f)) ? 100.f : (5.f - x0);
    float wallin[4] = { x0 + 5.f, x1 + 5.f, 5.f - x1, rightdist };
    float o[4];
    #pragma unroll
    for (int k = 0; k < 4; ++k)
        o[k] = mlp(wallin[k], w2a, b2a, b2b, w2c, b2c0, a_ww);
    float f2x = o[0] - o[3];
    float f2y = o[1] - o[2];

    // ---- f1: door-direction net 4->8->2 (pure VALU, lane-local) ----
    float dx = 5.6f - x0, dy = -x1;
    float inv = rsqrtf(fmaf(dx, dx, dy * dy));
    float in4[4] = { dx * inv, dy * inv, vx, vy };
    float hb[8];
    #pragma unroll
    for (int j = 0; j < 8; ++j) {
        float t = b1a[j];
        #pragma unroll
        for (int i = 0; i < 4; ++i) t = fmaf(in4[i], w1a[i*8 + j], t);
        hb[j] = fmaxf(t, 0.f);
    }
    float f1x = b1b[0], f1y = b1b[1];
    #pragma unroll
    for (int j = 0; j < 8; ++j) {
        f1x = fmaf(hb[j], w1b[j*2 + 0], f1x);
        f1y = fmaf(hb[j], w1b[j*2 + 1], f1y);
    }

    // ---- combine + escaped-particle mask + store (44 floats, 40 zeros) ----
    float dvx = (f1x + f2x + f3x) * 0.0125f;    // /80
    float dvy = (f1y + f2y + f3y) * 0.0125f;
    bool zrow = (x0 > 5.f);
    float4 head = zrow ? make_float4(0.f, 0.f, 0.f, 0.f)
                       : make_float4(vx, vy, dvx, dvy);
    if (tid < n) {
        float4* o4 = reinterpret_cast<float4*>(out);
        o4[row + 0] = head;
        float4 z = make_float4(0.f, 0.f, 0.f, 0.f);
        #pragma unroll
        for (int q = 1; q < 11; ++q) o4[row + q] = z;
    }
}

extern "C" void kernel_launch(void* const* d_in, const int* in_sizes, int n_in,
                              void* d_out, int out_size, void* d_ws, size_t ws_size,
                              hipStream_t stream) {
    // setup_inputs order: t, y, w1a, b1a, w1b, b1b, w2a, b2a, w2b, b2b, w2c,
    //                     b2c, w3a, b3a, w3b, b3b, w3c, b3c
    const float* y   = (const float*)d_in[1];
    const float* w1a = (const float*)d_in[2];
    const float* b1a = (const float*)d_in[3];
    const float* w1b = (const float*)d_in[4];
    const float* b1b = (const float*)d_in[5];
    const float* w2a = (const float*)d_in[6];
    const float* b2a = (const float*)d_in[7];
    const float* w2b = (const float*)d_in[8];
    const float* b2b = (const float*)d_in[9];
    const float* w2c = (const float*)d_in[10];
    const float* b2c = (const float*)d_in[11];
    const float* w3a = (const float*)d_in[12];
    const float* b3a = (const float*)d_in[13];
    const float* w3b = (const float*)d_in[14];
    const float* b3b = (const float*)d_in[15];
    const float* w3c = (const float*)d_in[16];
    const float* b3c = (const float*)d_in[17];

    int n = in_sizes[1] / 44;                   // N particles
    dim3 block(256);
    dim3 grid((n + 255) / 256);
    odefunc_kernel<<<grid, block, 0, stream>>>(
        y, w1a, b1a, w1b, b1b, w2a, b2a, w2b, b2b, w2c, b2c,
        w3a, b3a, w3b, b3b, w3c, b3c, (float*)d_out, n);
}
